// ModelDIN_66357244723531
// MI455X (gfx1250) — matrix operand order repaired and checked by reading, not compiled, not run on hardware
//
#include <hip/hip_runtime.h>
#include <hip/hip_bf16.h>

typedef __attribute__((ext_vector_type(16))) _Float16 v16h;
typedef __attribute__((ext_vector_type(8)))  _Float16 v8h;
typedef __attribute__((ext_vector_type(4)))  _Float16 v4h;
typedef __attribute__((ext_vector_type(8)))  float    v8f;

#define B_SZ 2048
#define T_SZ 200
#define TP   208     // T padded to 13 * 16 rows
#define H_SZ 128
#define E_SZ 64
#define NTHREADS 256

// ---- shared memory layout (single block, phase-overlapped) ----
// [0      , 53248) : hLDS   f16 [208][128]      (persistent)
// [53248  , 57344) : WbT    f16 [16][128]       (phase 1: fused attention weight)
// [53248  , 59904) : x1     f16 [208][16]       (phase 2: overlaps WbT after barrier)
// [59904  , 60416) : ieS    f32 [128]
// [60416  , 61248) : hIdx   i32 [208]
// [61248  , 61312) : qpart  f32 [16]
// [61312  , 62144) : scores f32 [208]
// [62144  , 62400) : red    f32 [64]
// [62400  , 63424) : hacc   f32 [2][128]
// [63424  , 64448) : din    f32 [256]
// [64448  , 64512) : o1s    f32 [16]
#define SMEM_BYTES 64512

// fast sigmoid: v_exp_f32 + v_add + v_rcp_f32 (avoids IEEE divide expansion)
__device__ __forceinline__ float sigmoidf(float x) {
    return __builtin_amdgcn_rcpf(1.0f + __expf(-x));
}

union HalfVec { v16h v; v8h h[2]; };

__global__ __launch_bounds__(NTHREADS) void din_fused_kernel(
    const int* __restrict__ iArr, const int* __restrict__ histArr,
    const int* __restrict__ slArr, const int* __restrict__ catList,
    const float* __restrict__ itemEmb, const float* __restrict__ catEmb,
    const float* __restrict__ aw1, const float* __restrict__ ab1,
    const float* __restrict__ aw2, const float* __restrict__ ab2,
    const float* __restrict__ aw3, const float* __restrict__ ab3,
    const float* __restrict__ aw4, const float* __restrict__ ab4,
    const float* __restrict__ bng, const float* __restrict__ bnb,
    const float* __restrict__ bnm, const float* __restrict__ bnv,
    const float* __restrict__ w1, const float* __restrict__ b1,
    const float* __restrict__ w2, const float* __restrict__ b2,
    const float* __restrict__ w3, const float* __restrict__ b3,
    float* __restrict__ outs)
{
    __shared__ __align__(64) unsigned char smem[SMEM_BYTES];
    _Float16* hLDS   = (_Float16*)(smem + 0);
    _Float16* WbT    = (_Float16*)(smem + 53248);
    _Float16* x1     = (_Float16*)(smem + 53248);   // overlaps WbT (phase-separated)
    float*    ieS    = (float*)   (smem + 59904);
    int*      hIdx   = (int*)     (smem + 60416);
    float*    qpart  = (float*)   (smem + 61248);
    float*    scores = (float*)   (smem + 61312);
    float*    red    = (float*)   (smem + 62144);
    float*    hacc   = (float*)   (smem + 62400);
    float*    din    = (float*)   (smem + 63424);
    float*    o1s    = (float*)   (smem + 64448);

    const int b    = blockIdx.x;
    const int tid  = threadIdx.x;
    const int lane = tid & 31;
    const int wave = tid >> 5;

    const int iIdx  = iArr[b];
    const int icIdx = catList[iIdx];
    const int slb   = slArr[b];

    const float4* item4 = (const float4*)itemEmb;
    const float4* cat4  = (const float4*)catEmb;

    // ---------------- phase 1: i_e and history indices ----------------
    if (tid < 32) {
        int k4 = tid;
        float4 v = (k4 < 16) ? item4[iIdx * 16 + k4] : cat4[icIdx * 16 + (k4 - 16)];
        ieS[k4 * 4 + 0] = v.x; ieS[k4 * 4 + 1] = v.y;
        ieS[k4 * 4 + 2] = v.z; ieS[k4 * 4 + 3] = v.w;
    }
    if (tid < TP) hIdx[tid] = (tid < T_SZ) ? histArr[b * T_SZ + tid] : 0;
    __syncthreads();

    // ---------------- phase 2: gather h_e (f16), fused weight WbT, qpart ----------------
    // h_e[t] = [item_emb[hist], cat_emb[hist]]  -> hLDS[t][0..127], rows >= T zeroed
    for (int idx = tid; idx < TP * 32; idx += NTHREADS) {
        int t = idx >> 5, k4 = idx & 31;
        float4 v = make_float4(0.f, 0.f, 0.f, 0.f);
        if (t < T_SZ) {
            int hi = hIdx[t];
            v = (k4 < 16) ? item4[hi * 16 + k4] : cat4[hi * 16 + (k4 - 16)];
        }
        v4h hv = { (_Float16)v.x, (_Float16)v.y, (_Float16)v.z, (_Float16)v.w };
        *(v4h*)(hLDS + t * H_SZ + k4 * 4) = hv;
    }
    // Wb[k][j] = aw1[H+k][j] - aw1[2H+k][j] + q[k]*aw1[3H+k][j]  (stored transposed [j][k])
    for (int idx = tid; idx < 16 * H_SZ; idx += NTHREADS) {
        int j = idx >> 7, k = idx & 127;
        float w = aw1[(H_SZ + k) * 16 + j] - aw1[(2 * H_SZ + k) * 16 + j]
                + ieS[k] * aw1[(3 * H_SZ + k) * 16 + j];
        WbT[j * H_SZ + k] = (_Float16)w;
    }
    // qpart[j] = ab1[j] + sum_k q[k]*(aw1[k][j] + aw1[2H+k][j])
    if (tid < 16) {
        int j = tid;
        float q = ab1[j];
        for (int k = 0; k < H_SZ; ++k)
            q += ieS[k] * (aw1[k * 16 + j] + aw1[(2 * H_SZ + k) * 16 + j]);
        qpart[j] = q;
    }
    __syncthreads();

    // ---------------- phase 3: load B fragments (K=32 steps x4), then free WbT ----------------
    const int col   = lane & 15;
    const int off8  = (lane >> 4) * 8;    // A-matrix K sub-chunk select
    const int off16 = (lane >> 4) * 16;   // unused: B is loaded with off8, same k order as A
    v16h bfrag[4];
    #pragma unroll
    for (int ks = 0; ks < 4; ++ks)
    {
        HalfVec bt;
        bt.h[0] = *(const v8h*)(WbT + col * H_SZ + ks * 32 + off8);
        bt.h[1] = *(const v8h*)(WbT + col * H_SZ + ks * 32 + 16 + off8);
        bfrag[ks] = bt.v;
    }
    __syncthreads();   // after this, WbT region is reused as x1

    // ---------------- phase 4: WMMA  x1 = sigmoid(h[208x128] @ Wb[128x16] + qpart) ----------
    const float qp = qpart[col];
    for (int tile = wave; tile < TP / 16; tile += (NTHREADS / 32)) {
        const int row = tile * 16 + (lane & 15);
        v8f c;
        #pragma unroll
        for (int r = 0; r < 8; ++r) c[r] = qp;
        #pragma unroll
        for (int ks = 0; ks < 4; ++ks) {
            HalfVec a;
            const _Float16* ap = hLDS + row * H_SZ + ks * 32 + off8;
            a.h[0] = *(const v8h*)(ap);        // K = ks*32 + off8 + [0..7]
            a.h[1] = *(const v8h*)(ap + 16);   // K = ks*32 + 16 + off8 + [0..7]
            c = __builtin_amdgcn_wmma_f32_16x16x32_f16(
                    false, a.v, false, bfrag[ks], (short)0, c, false, false);
        }
        #pragma unroll
        for (int r = 0; r < 8; ++r) {
            int rr = tile * 16 + r + ((lane >> 4) << 3);
            x1[rr * 16 + col] = (_Float16)sigmoidf(c[r]);
        }
    }
    __syncthreads();

    // ---------------- phase 5: tiny MLP layers 2/3 + mask ----------------
    if (tid < TP) {
        const float NEGMASK = -4294967295.0f;            // -(2^32)+1
        const float RSQ     = 0.08838834764831845f;      // 1/sqrt(128)
        float s;
        if (tid < T_SZ) {
            float xr[16];
            #pragma unroll
            for (int j = 0; j < 16; ++j) xr[j] = (float)x1[tid * 16 + j];
            float o2[8];
            #pragma unroll
            for (int m = 0; m < 8; ++m) {
                float a = ab2[m];
                #pragma unroll
                for (int j = 0; j < 16; ++j) a += xr[j] * aw2[j * 8 + m];
                o2[m] = sigmoidf(a);
            }
            s = ab3[0];
            #pragma unroll
            for (int m = 0; m < 8; ++m) s += o2[m] * aw3[m];
            s = (tid < slb) ? s : NEGMASK;
            s *= RSQ;
        } else {
            s = NEGMASK * RSQ;
        }
        scores[tid] = s;
    }
    __syncthreads();

    // ---------------- phase 6: softmax over 208 (block reduction) ----------------
    {
        float val = (tid < TP) ? scores[tid] : -3.0e38f;
        float m = val;
        #pragma unroll
        for (int o = 16; o > 0; o >>= 1) m = fmaxf(m, __shfl_xor(m, o, 32));
        if (lane == 0) red[wave] = m;
        __syncthreads();
        if (tid == 0) {
            float mm = red[0];
            for (int r = 1; r < 8; ++r) mm = fmaxf(mm, red[r]);
            red[8] = mm;
        }
        __syncthreads();
        float gmax = red[8];
        float e = __expf(val - gmax);
        float ssum = e;
        #pragma unroll
        for (int o = 16; o > 0; o >>= 1) ssum += __shfl_xor(ssum, o, 32);
        if (lane == 0) red[16 + wave] = ssum;
        __syncthreads();
        if (tid == 0) {
            float t = 0.f;
            for (int r = 0; r < 8; ++r) t += red[16 + r];
            red[9] = t;
        }
        __syncthreads();
        float inv = __builtin_amdgcn_rcpf(red[9]);
        if (tid < TP) scores[tid] = e * inv;   // scores now holds attention
    }
    __syncthreads();

    // ---------------- phase 7: hist = att @ h_e ; fc4 ; BN ; head MLP -----------
    {
        int k = tid & 127, half = tid >> 7;
        float acc = 0.f;
        int t0 = half * (TP / 2);
        for (int t = t0; t < t0 + TP / 2; ++t)
            acc += scores[t] * (float)hLDS[t * H_SZ + k];
        hacc[half * H_SZ + k] = acc;
    }
    __syncthreads();
    if (tid < H_SZ) hacc[tid] = hacc[tid] + hacc[H_SZ + tid];
    __syncthreads();
    if (tid < H_SZ) {
        int j = tid;
        float a = ab4[j];
        for (int k = 0; k < H_SZ; ++k) a += hacc[k] * aw4[k * H_SZ + j];
        float d0 = (a - bnm[j]) * __builtin_amdgcn_rsqf(bnv[j] + 1e-3f) * bng[j] + bnb[j];
        int j2 = H_SZ + j;
        float d1 = (ieS[j] - bnm[j2]) * __builtin_amdgcn_rsqf(bnv[j2] + 1e-3f) * bng[j2] + bnb[j2];
        din[j] = d0;
        din[H_SZ + j] = d1;
    }
    __syncthreads();
    if (tid < 16) {
        float a = b1[tid];
        for (int k = 0; k < 2 * H_SZ; ++k) a += din[k] * w1[k * 16 + tid];
        o1s[tid] = a;
    }
    __syncthreads();
    if (tid == 0) {
        float o2v[8];
        #pragma unroll
        for (int m = 0; m < 8; ++m) o2v[m] = b2[m];
        for (int k = 0; k < 16; ++k) {
            float v = o1s[k];
            #pragma unroll
            for (int m = 0; m < 8; ++m) o2v[m] += v * w2[k * 8 + m];
        }
        float o3 = b3[0];
        #pragma unroll
        for (int m = 0; m < 8; ++m) o3 += o2v[m] * w3[m];
        outs[b] = o3;
    }
}

// ibv[c] = item_b[i[c]]
__global__ __launch_bounds__(256) void din_ib_kernel(
    const int* __restrict__ iArr, const float* __restrict__ itemB,
    float* __restrict__ ibv)
{
    int idx = blockIdx.x * 256 + threadIdx.x;
    if (idx < B_SZ) ibv[idx] = itemB[iArr[idx]];
}

// out[r, c] = outs[r] + ibv[c]   (TF broadcast [B] + [B,1] -> [B,B])
__global__ __launch_bounds__(256) void din_bcast_kernel(
    const float* __restrict__ outs, const float* __restrict__ ibv,
    float* __restrict__ out)
{
    int idx = blockIdx.x * 256 + threadIdx.x;
    int r = idx >> 11;        // / 2048
    int c = idx & (B_SZ - 1); // % 2048
    out[idx] = outs[r] + ibv[c];
}

extern "C" void kernel_launch(void* const* d_in, const int* in_sizes, int n_in,
                              void* d_out, int out_size, void* d_ws, size_t ws_size,
                              hipStream_t stream) {
    (void)in_sizes; (void)n_in; (void)out_size; (void)ws_size;
    // input order from setup_inputs():
    // 0 u, 1 i, 2 hist_i, 3 sl, 4 category_list, 5 item_emb, 6 cat_emb, 7 item_b,
    // 8 aw1, 9 ab1, 10 aw2, 11 ab2, 12 aw3, 13 ab3, 14 aw4, 15 ab4,
    // 16 bn_gamma, 17 bn_beta, 18 bn_mean, 19 bn_var,
    // 20 w1, 21 b1, 22 w2, 23 b2, 24 w3, 25 b3
    const int*   iArr    = (const int*)  d_in[1];
    const int*   histArr = (const int*)  d_in[2];
    const int*   slArr   = (const int*)  d_in[3];
    const int*   catList = (const int*)  d_in[4];
    const float* itemEmb = (const float*)d_in[5];
    const float* catEmb  = (const float*)d_in[6];
    const float* itemB   = (const float*)d_in[7];
    const float* aw1 = (const float*)d_in[8];
    const float* ab1 = (const float*)d_in[9];
    const float* aw2 = (const float*)d_in[10];
    const float* ab2 = (const float*)d_in[11];
    const float* aw3 = (const float*)d_in[12];
    const float* ab3 = (const float*)d_in[13];
    const float* aw4 = (const float*)d_in[14];
    const float* ab4 = (const float*)d_in[15];
    const float* bng = (const float*)d_in[16];
    const float* bnb = (const float*)d_in[17];
    const float* bnm = (const float*)d_in[18];
    const float* bnv = (const float*)d_in[19];
    const float* w1  = (const float*)d_in[20];
    const float* b1  = (const float*)d_in[21];
    const float* w2  = (const float*)d_in[22];
    const float* b2  = (const float*)d_in[23];
    const float* w3  = (const float*)d_in[24];
    const float* b3  = (const float*)d_in[25];

    float* outs = (float*)d_ws;          // [2048] per-row scalar outputs
    float* ibv  = outs + B_SZ;           // [2048] item_b[i[c]]
    float* out  = (float*)d_out;         // [2048 x 2048]

    din_fused_kernel<<<B_SZ, NTHREADS, 0, stream>>>(
        iArr, histArr, slArr, catList, itemEmb, catEmb,
        aw1, ab1, aw2, ab2, aw3, ab3, aw4, ab4,
        bng, bnb, bnm, bnv, w1, b1, w2, b2, w3, b3, outs);

    din_ib_kernel<<<(B_SZ + 255) / 256, 256, 0, stream>>>(iArr, itemB, ibv);

    din_bcast_kernel<<<(B_SZ * B_SZ) / 256, 256, 0, stream>>>(outs, ibv, out);
}
